// SignedDirectedAttention_87393994539668
// MI455X (gfx1250) — matrix operand order repaired and checked by reading, not compiled, not run on hardware
//
#include <hip/hip_runtime.h>

// Problem constants (fixed by the reference): N=50000, D=64, H=4, R=4, E=800000.
#define DIM   64
#define HEADS 4
#define RELS  4
#define HD    256    // HEADS*DIM
#define RHD   1024   // RELS*HEADS*DIM

typedef __bf16 v16bf __attribute__((ext_vector_type(16)));
typedef float  v8f   __attribute__((ext_vector_type(8)));

// Monotonic f32 <-> u32 mapping so segment-max can use hardware u32 atomicMax.
__device__ __forceinline__ unsigned f32_to_ordered(float f) {
  unsigned u = __float_as_uint(f);
  return (u & 0x80000000u) ? ~u : (u | 0x80000000u);
}
__device__ __forceinline__ float ordered_to_f32(unsigned k) {
  unsigned u = (k & 0x80000000u) ? (k & 0x7fffffffu) : ~k;
  return __uint_as_float(u);
}

// ---------------------------------------------------------------------------
// Pack a weight matrix W[32*nkc, ncols] into WMMA B-fragment order, bf16x2
// (hi then lo), so the GEMM inner loop does two contiguous b128 loads per
// operand and no conversions.  Fragment f = tile*nkc + kc; per-lane payload is
// 16 hi bf16 + 16 lo bf16 = 64 contiguous bytes at P + (f*32 + lane)*32 elems.
// One wave packs one fragment.
// ---------------------------------------------------------------------------
__global__ void sda_pack_w(const float* __restrict__ W, __bf16* __restrict__ P,
                           int ncols, int nkc) {
  const int f    = blockIdx.x * 8 + (threadIdx.x >> 5);
  const int lane = threadIdx.x & 31;
  const int tile = f / nkc;
  const int kc   = f % nkc;
  const int l15  = lane & 15;
  const int hi   = lane >> 4;
  const int col  = tile * 16 + l15;
  const int kb   = kc * 32 + hi * 8;             // B layout: lanes 0-15 K {0..7, 16..23}; lanes 16-31 +8
  __bf16* dst = P + ((size_t)f * 32 + lane) * 32;
#pragma unroll
  for (int i = 0; i < 16; ++i) {
    float wval = W[(size_t)(kb + i + (i >= 8 ? 8 : 0)) * ncols + col];
    __bf16 h = (__bf16)wval;
    dst[i]      = h;
    dst[16 + i] = (__bf16)(wval - (float)h);
  }
}

// Build one A fragment pair (hi/lo) for rows [row0..row0+15], k-chunk base kc*32.
__device__ __forceinline__ void sda_load_a(const float* __restrict__ arow,
                                           int kc, int hi,
                                           v16bf& ah, v16bf& al) {
  // ISA 16-bit A layout: lanes 0-15 hold K {kc*32 + 0..7, +16..23}; lanes 16-31 +8.
  const float4* xa = (const float4*)(arow + kc * 32 + hi * 8);
  float xv[16];
  *(float4*)(xv + 0)  = xa[0];
  *(float4*)(xv + 4)  = xa[1];
  *(float4*)(xv + 8)  = xa[4];   // +16 floats
  *(float4*)(xv + 12) = xa[5];
#pragma unroll
  for (int i = 0; i < 16; ++i) {
    __bf16 h = (__bf16)xv[i];
    ah[i] = h;
    al[i] = (__bf16)(xv[i] - (float)h);
  }
}

// ---------------------------------------------------------------------------
// Y[N,256] = X[N,64] @ W[64,256] + b   via v_wmma_f32_16x16x32_bf16, bf16x2
// split (hi*hi + hi*lo + lo*hi) for ~fp32 accuracy.  Block = 256 (8 waves);
// each WAVE owns one 16-row stripe and sweeps all 16 column tiles, so the
// A-fragment load+split happens once per stripe and feeds 96 WMMAs.
// ---------------------------------------------------------------------------
__global__ void sda_qkv_gemm(const float* __restrict__ X,
                             const __bf16* __restrict__ Wpk,
                             const float* __restrict__ b,
                             float* __restrict__ Y, int nStripes) {
  const int lane   = threadIdx.x & 31;
  const int wv     = threadIdx.x >> 5;
  const int stripe = blockIdx.x * 8 + wv;
  if (stripe >= nStripes) return;
  const int row0 = stripe * 16;
  const int l15  = lane & 15;
  const int hi   = lane >> 4;

  v16bf ah[2], al[2];
  const float* __restrict__ arow = X + (size_t)(row0 + l15) * DIM;
  sda_load_a(arow, 0, hi, ah[0], al[0]);
  sda_load_a(arow, 1, hi, ah[1], al[1]);

  for (int tile = 0; tile < 16; ++tile) {
    const int col  = tile * 16 + l15;
    const float bias = b[col];
    v8f acc;
#pragma unroll
    for (int j = 0; j < 8; ++j) acc[j] = bias;  // bias depends on column only

#pragma unroll
    for (int kc = 0; kc < 2; ++kc) {
      const v16bf* bp =
          (const v16bf*)(Wpk + ((size_t)(tile * 2 + kc) * 32 + lane) * 32);
      v16bf bh = bp[0];
      v16bf bl = bp[1];
      acc = __builtin_amdgcn_wmma_f32_16x16x32_bf16(false, ah[kc], false, bh,
                                                    (short)0, acc, false, false);
      acc = __builtin_amdgcn_wmma_f32_16x16x32_bf16(false, ah[kc], false, bl,
                                                    (short)0, acc, false, false);
      acc = __builtin_amdgcn_wmma_f32_16x16x32_bf16(false, al[kc], false, bh,
                                                    (short)0, acc, false, false);
    }

#pragma unroll
    for (int j = 0; j < 8; ++j) {
      const int m = j + hi * 8;                 // C/D layout: VGPR j -> M=j / j+8
      Y[(size_t)(row0 + m) * HD + col] = acc[j];
    }
  }
}

// ---------------------------------------------------------------------------
// Edge scores + segment max.  One wave per edge: 8 lanes per head, each lane
// covers 8 contiguous floats of the 64-wide head dot, shfl_xor tree reduce.
// ---------------------------------------------------------------------------
__global__ void sda_edge_score(const float* __restrict__ Q,
                               const float* __restrict__ K,
                               const int* __restrict__ src,
                               const int* __restrict__ tgt,
                               const float* __restrict__ nsi,
                               const float* __restrict__ sw,   // [H,R]
                               int r, float* __restrict__ scores,
                               unsigned* __restrict__ mkey, int E) {
  const int e = blockIdx.x * 8 + (threadIdx.x >> 5);
  if (e >= E) return;
  const int lane = threadIdx.x & 31;
  const int s = src[e], t = tgt[e];
  const int h = lane >> 3;
  const int o = (lane & 7) * 8;

  const float4* qp = (const float4*)(Q + (size_t)s * HD + h * DIM + o);
  const float4* kp = (const float4*)(K + (size_t)t * HD + h * DIM + o);
  float4 q0 = qp[0], q1 = qp[1];
  float4 k0 = kp[0], k1 = kp[1];
  float p = q0.x * k0.x + q0.y * k0.y + q0.z * k0.z + q0.w * k0.w
          + q1.x * k1.x + q1.y * k1.y + q1.z * k1.z + q1.w * k1.w;
  p += __shfl_xor(p, 1, 32);
  p += __shfl_xor(p, 2, 32);
  p += __shfl_xor(p, 4, 32);                     // all 8 lanes of a head agree

  const float sc = p * 0.125f * nsi[s] * sw[h * RELS + r];  // 1/sqrt(64)
  if ((lane & 7) == 0) {
    scores[(size_t)e * HEADS + h] = sc;
    atomicMax(&mkey[(size_t)s * HEADS + h], f32_to_ordered(sc));
  }
}

// ---------------------------------------------------------------------------
// e = exp(score - m[src]); segment-sum into ssum; scores overwritten with e.
// One thread per edge handles all 4 heads (float4 traffic).
// ---------------------------------------------------------------------------
__global__ void sda_exp_sum(float* __restrict__ scores,
                            const int* __restrict__ src,
                            const unsigned* __restrict__ mkey,
                            float* __restrict__ ssum, int E) {
  const int e = blockIdx.x * blockDim.x + threadIdx.x;
  if (e >= E) return;
  const int s = src[e];
  float4 sc = *(const float4*)(scores + (size_t)e * HEADS);
  uint4  mk = *(const uint4*)(mkey + (size_t)s * HEADS);
  float e0 = __expf(sc.x - ordered_to_f32(mk.x));
  float e1 = __expf(sc.y - ordered_to_f32(mk.y));
  float e2 = __expf(sc.z - ordered_to_f32(mk.z));
  float e3 = __expf(sc.w - ordered_to_f32(mk.w));
  *(float4*)(scores + (size_t)e * HEADS) = make_float4(e0, e1, e2, e3);
  float* sp = ssum + (size_t)s * HEADS;
  atomicAdd(sp + 0, e0);
  atomicAdd(sp + 1, e1);
  atomicAdd(sp + 2, e2);
  atomicAdd(sp + 3, e3);
}

// ---------------------------------------------------------------------------
// Scatter-add of attn * V[tgt] into combined[src, r*256 + h*64 + d].
// One wave per edge; 8 global f32 atomic adds per lane (L2-resident target).
// ---------------------------------------------------------------------------
__global__ void sda_aggregate(const float* __restrict__ scores,  // holds e
                              const float* __restrict__ ssum,
                              const float* __restrict__ V,
                              const int* __restrict__ src,
                              const int* __restrict__ tgt,
                              float* __restrict__ combR, int E) {
  const int e = blockIdx.x * 8 + (threadIdx.x >> 5);
  if (e >= E) return;
  const int lane = threadIdx.x & 31;
  const int s = src[e], t = tgt[e];
  const int h = lane >> 3;
  const int o = (lane & 7) * 8;

  const float attn = scores[(size_t)e * HEADS + h] /
                     (ssum[(size_t)s * HEADS + h] + 1e-10f);
  const float4* vp = (const float4*)(V + (size_t)t * HD + h * DIM + o);
  float4 v0 = vp[0], v1 = vp[1];
  float* op = combR + (size_t)s * RHD + h * DIM + o;
  atomicAdd(op + 0, attn * v0.x);
  atomicAdd(op + 1, attn * v0.y);
  atomicAdd(op + 2, attn * v0.z);
  atomicAdd(op + 3, attn * v0.w);
  atomicAdd(op + 4, attn * v1.x);
  atomicAdd(op + 5, attn * v1.y);
  atomicAdd(op + 6, attn * v1.z);
  atomicAdd(op + 7, attn * v1.w);
}

// ---------------------------------------------------------------------------
// out[N,64] = combined[N,1024] @ Wo[1024,64] + bo  (bf16x2 WMMA, K=32 chunks,
// pre-packed B fragments: 4 tiles x 32 k-chunks).  Block = 256 (8 waves);
// each WAVE owns one 16-row stripe and keeps all four 16-col accumulators
// live, so each A-fragment split feeds 12 WMMAs.
// ---------------------------------------------------------------------------
__global__ void sda_out_gemm(const float* __restrict__ C,
                             const __bf16* __restrict__ Wpk,
                             const float* __restrict__ bo,
                             float* __restrict__ out, int nStripes) {
  const int lane   = threadIdx.x & 31;
  const int wv     = threadIdx.x >> 5;
  const int stripe = blockIdx.x * 8 + wv;
  if (stripe >= nStripes) return;
  const int row0 = stripe * 16;
  const int l15  = lane & 15;
  const int hi   = lane >> 4;

  v8f acc[4];
#pragma unroll
  for (int t = 0; t < 4; ++t) {
    const float bias = bo[t * 16 + l15];
#pragma unroll
    for (int j = 0; j < 8; ++j) acc[t][j] = bias;
  }

  const float* __restrict__ arow = C + (size_t)(row0 + l15) * RHD;
  for (int kc = 0; kc < 32; ++kc) {
    v16bf ah, al;
    sda_load_a(arow, kc, hi, ah, al);
#pragma unroll
    for (int t = 0; t < 4; ++t) {
      const v16bf* bp =
          (const v16bf*)(Wpk + ((size_t)(t * 32 + kc) * 32 + lane) * 32);
      v16bf bh = bp[0];
      v16bf bl = bp[1];
      acc[t] = __builtin_amdgcn_wmma_f32_16x16x32_bf16(false, ah, false, bh,
                                                       (short)0, acc[t], false, false);
      acc[t] = __builtin_amdgcn_wmma_f32_16x16x32_bf16(false, ah, false, bl,
                                                       (short)0, acc[t], false, false);
      acc[t] = __builtin_amdgcn_wmma_f32_16x16x32_bf16(false, al, false, bh,
                                                       (short)0, acc[t], false, false);
    }
  }

#pragma unroll
  for (int t = 0; t < 4; ++t) {
#pragma unroll
    for (int j = 0; j < 8; ++j) {
      const int m = j + hi * 8;
      out[(size_t)(row0 + m) * DIM + t * 16 + l15] = acc[t][j];
    }
  }
}

// ---------------------------------------------------------------------------
extern "C" void kernel_launch(void* const* d_in, const int* in_sizes, int n_in,
                              void* d_out, int out_size, void* d_ws, size_t ws_size,
                              hipStream_t stream) {
  (void)n_in; (void)out_size; (void)ws_size;
  const float* X    = (const float*)d_in[0];
  const float* nsi  = (const float*)d_in[1];
  const int*   esrc = (const int*)  d_in[2];
  const int*   etgt = (const int*)  d_in[3];
  const float* Wq   = (const float*)d_in[4];
  const float* bq   = (const float*)d_in[5];
  const float* Wk   = (const float*)d_in[6];
  const float* bk   = (const float*)d_in[7];
  const float* Wv   = (const float*)d_in[8];
  const float* bv   = (const float*)d_in[9];
  const float* sw   = (const float*)d_in[10];
  const float* Wo   = (const float*)d_in[11];
  const float* bo   = (const float*)d_in[12];
  float* out = (float*)d_out;

  const int Nn = in_sizes[0] / DIM;        // 50000 (multiple of 16)
  const int E  = in_sizes[2] / RELS;       // 800000
  const int nStripes = Nn / 16;            // 3125

  // Workspace partition (relation-sequenced so the hot set stays in L2).
  char* p = (char*)d_ws;
  auto carve = [&](size_t bytes) {
    char* r = p;
    p += (bytes + 255) & ~(size_t)255;
    return r;
  };
  float*    Qb  = (float*)   carve((size_t)Nn * HD * 4);
  float*    Kb  = (float*)   carve((size_t)Nn * HD * 4);
  float*    Vb  = (float*)   carve((size_t)Nn * HD * 4);
  float*    Sc  = (float*)   carve((size_t)E * HEADS * 4);
  unsigned* Mk  = (unsigned*)carve((size_t)Nn * HEADS * 4);
  float*    Ss  = (float*)   carve((size_t)Nn * HEADS * 4);
  float*    Cb  = (float*)   carve((size_t)Nn * RHD * 4);
  // Packed B fragments: qkv W -> 16 tiles * 2 kc * 32 lanes * 32 bf16 (64 KB)
  __bf16*   Qpk = (__bf16*)  carve((size_t)16 * 2 * 32 * 32 * 2);
  __bf16*   Kpk = (__bf16*)  carve((size_t)16 * 2 * 32 * 32 * 2);
  __bf16*   Vpk = (__bf16*)  carve((size_t)16 * 2 * 32 * 32 * 2);
  // Wo -> 4 tiles * 32 kc * 32 lanes * 32 bf16 (256 KB)
  __bf16*   Opk = (__bf16*)  carve((size_t)4 * 32 * 32 * 32 * 2);

  hipMemsetAsync(Cb, 0, (size_t)Nn * RHD * 4, stream);
  sda_pack_w<<<dim3((4 * 32) / 8), 256, 0, stream>>>(Wo, Opk, DIM, 32);

  const dim3 gemmGrid((nStripes + 7) / 8);   // one wave per 16-row stripe
  const dim3 packGrid((16 * 2) / 8);         // 32 fragments, 8 waves/block
  const int edgeBlocks = (E + 7) / 8;        // one wave per edge
  const int thrBlocks  = (E + 255) / 256;    // one thread per edge

  for (int r = 0; r < RELS; ++r) {
    const size_t wOff = (size_t)r * DIM * HD;
    sda_pack_w<<<packGrid, 256, 0, stream>>>(Wq + wOff, Qpk, HD, 2);
    sda_pack_w<<<packGrid, 256, 0, stream>>>(Wk + wOff, Kpk, HD, 2);
    sda_pack_w<<<packGrid, 256, 0, stream>>>(Wv + wOff, Vpk, HD, 2);
    sda_qkv_gemm<<<gemmGrid, 256, 0, stream>>>(X, Qpk, bq + (size_t)r * HD, Qb, nStripes);
    sda_qkv_gemm<<<gemmGrid, 256, 0, stream>>>(X, Kpk, bk + (size_t)r * HD, Kb, nStripes);
    sda_qkv_gemm<<<gemmGrid, 256, 0, stream>>>(X, Vpk, bv + (size_t)r * HD, Vb, nStripes);

    hipMemsetAsync(Mk, 0, (size_t)Nn * HEADS * 4, stream);
    hipMemsetAsync(Ss, 0, (size_t)Nn * HEADS * 4, stream);

    const int* sr = esrc + (size_t)r * E;
    const int* tr = etgt + (size_t)r * E;
    sda_edge_score<<<edgeBlocks, 256, 0, stream>>>(Qb, Kb, sr, tr, nsi, sw, r, Sc, Mk, E);
    sda_exp_sum<<<thrBlocks, 256, 0, stream>>>(Sc, sr, Mk, Ss, E);
    sda_aggregate<<<edgeBlocks, 256, 0, stream>>>(Sc, Ss, Vb, sr, tr, Cb + (size_t)r * HD, E);
  }

  sda_out_gemm<<<gemmGrid, 256, 0, stream>>>(Cb, Opk, bo, out, nStripes);
}
